// E74DiagonalCell_23751169147414
// MI455X (gfx1250) — matrix operand order repaired and checked by reading, not compiled, not run on hardware
//
#include <hip/hip_runtime.h>
#include <cmath>

// ---------------------------------------------------------------------------
// Problem constants (match reference): T=2048, B=16, DIM=1024, N_STATE=1024
// ---------------------------------------------------------------------------
constexpr int T_   = 2048;
constexpr int B_   = 16;
constexpr int DIM_ = 1024;
constexpr int NST  = 1024;
constexpr int M_TOTAL = T_ * B_;            // 32768 rows for the projections
constexpr float EPS_ = 1e-6f;

typedef __attribute__((ext_vector_type(16))) __bf16 v16bf;
typedef __attribute__((ext_vector_type(8)))  __bf16 v8bf;
typedef __attribute__((ext_vector_type(8)))  float  v8f;

// GEMM tiling
constexpr int BM = 128, BN = 128, BK = 32;
constexpr int NKB = DIM_ / BK;               // 32 k-steps
constexpr int FROW = BK + 4;                 // f32 staging row stride (144B, 16B aligned)
constexpr int LDB  = BK + 8;                 // bf16 row stride (80B, 16B aligned)
constexpr int FBUF = BM * FROW;              // floats per staging buffer per matrix

__device__ __forceinline__ v8f wmma_bf16(v16bf a, v16bf b, v8f c) {
  // 8 args: (neg_a, A, neg_b, B, c_mod, C, reuse_a, reuse_b)
  return __builtin_amdgcn_wmma_f32_16x16x32_bf16(false, a, false, b, (short)0, c,
                                                 false, false);
}

__device__ __forceinline__ v16bf concat8(v8bf lo, v8bf hi) {
  return __builtin_shufflevector(lo, hi, 0, 1, 2, 3, 4, 5, 6, 7,
                                         8, 9, 10, 11, 12, 13, 14, 15);
}

// Async global -> LDS copy of 16 bytes (per lane). ASYNCcnt-tracked.
__device__ __forceinline__ void async_b128(unsigned lds_off, const float* g) {
  asm volatile("global_load_async_to_lds_b128 %0, %1, off"
               :: "v"(lds_off), "v"((unsigned long long)(uintptr_t)g)
               : "memory");
}

// ---------------------------------------------------------------------------
// Kernel 1: C[m,n] = sum_d X[m,d] * W[n,d]   for W in {W_k, W_v, W_q}
// Split-bf16: X = Xhi+Xlo, W = Whi+Wlo; acc += Xhi*Whi + Xhi*Wlo + Xlo*Whi.
// Double-buffered fp32 staging in LDS via global_load_async_to_lds_b128; the
// hi/lo bf16 split is done LDS->LDS once per tile (no per-fragment dup work).
// Block: 256 threads = 8 waves; block tile 128x128, wave tile 32(m) x 64(n).
// blockIdx.y in [0,24): [0,8)->W_k, [8,16)->W_v, [16,24)->W_q.
// ---------------------------------------------------------------------------
__global__ __launch_bounds__(256, 1)
void gemm3_kernel(const float* __restrict__ X,
                  const float* __restrict__ Wk,
                  const float* __restrict__ Wv,
                  const float* __restrict__ Wq,
                  float* __restrict__ Kout,
                  float* __restrict__ Vout,
                  float* __restrict__ Qout) {
  extern __shared__ __align__(16) char smem[];
  float*  Xf  = (float*)smem;                       // [2][BM][FROW] fp32 staging (X)
  float*  Wf  = Xf + 2 * FBUF;                      // [2][BM][FROW] fp32 staging (W)
  __bf16* Ahi = (__bf16*)(Wf + 2 * FBUF);           // [BM][LDB]
  __bf16* Alo = Ahi + BM * LDB;
  __bf16* Bh  = Alo + BM * LDB;
  __bf16* Bl  = Bh  + BM * LDB;

  const unsigned ldsXf = (unsigned)(uintptr_t)Xf;   // LDS byte offsets for async
  const unsigned ldsWf = (unsigned)(uintptr_t)Wf;

  const int tid   = threadIdx.x;
  const int mBase = blockIdx.x * BM;
  const int nb    = blockIdx.y;
  const int matId = nb >> 3;                        // 0:k 1:v 2:q
  const int nBase = (nb & 7) * BN;

  const float* __restrict__ Wp = (matId == 0) ? Wk : (matId == 1) ? Wv : Wq;
  float* __restrict__       Cp = (matId == 0) ? Kout : (matId == 1) ? Vout : Qout;

  const float* __restrict__ Xg = X  + (size_t)mBase * DIM_;
  const float* __restrict__ Wg = Wp + (size_t)nBase * DIM_;

  const int lane  = tid & 31;
  const int wave  = tid >> 5;
  const int waveM = (wave & 3) * 32;                // 4 waves along m -> 128
  const int waveN = (wave >> 2) * 64;               // 2 waves along n -> 128
  const int l16   = lane & 15;
  const int hsel  = lane >> 4;                      // 0: lanes 0-15, 1: lanes 16-31

  // per-thread staging coordinates: 4 float4 chunks per matrix per tile
  int srow[4], scol[4];
  #pragma unroll
  for (int i = 0; i < 4; ++i) {
    const int lin = tid + i * 256;                  // float4 index 0..1023
    srow[i] = lin >> 3;                             // 0..127
    scol[i] = (lin & 7) * 4;                        // 0,4,...,28
  }

  // issue async loads of tile `kb` into staging buffer `buf`
  auto stage = [&](int kb, int buf) {
    const unsigned xb = ldsXf + (unsigned)(buf * FBUF * 4);
    const unsigned wb = ldsWf + (unsigned)(buf * FBUF * 4);
    #pragma unroll
    for (int i = 0; i < 4; ++i) {
      const unsigned lo = (unsigned)((srow[i] * FROW + scol[i]) * 4);
      async_b128(xb + lo, Xg + (size_t)srow[i] * DIM_ + kb * BK + scol[i]);
      async_b128(wb + lo, Wg + (size_t)srow[i] * DIM_ + kb * BK + scol[i]);
    }
  };

  v8f acc[8];                                       // [mi*4+ni]
  #pragma unroll
  for (int i = 0; i < 8; ++i) acc[i] = (v8f)0.0f;

  stage(0, 0);                                      // prologue

  for (int kb = 0; kb < NKB; ++kb) {
    const int cur = kb & 1;
    if (kb + 1 < NKB) {
      stage(kb + 1, cur ^ 1);
      asm volatile("s_wait_asynccnt 0x8" ::: "memory");  // tile `kb` complete
    } else {
      asm volatile("s_wait_asynccnt 0x0" ::: "memory");
    }
    __syncthreads();                                // all waves' tile kb visible

    // ---- LDS fp32 -> LDS bf16 hi/lo split (8192 converts per block tile) ----
    {
      const float* xf = Xf + cur * FBUF;
      const float* wf = Wf + cur * FBUF;
      #pragma unroll
      for (int i = 0; i < 4; ++i) {
        const int r = srow[i], c = scol[i];
        const float4 xv = *(const float4*)(xf + r * FROW + c);
        const float4 wv = *(const float4*)(wf + r * FROW + c);
        const float xs[4] = {xv.x, xv.y, xv.z, xv.w};
        const float ws4[4] = {wv.x, wv.y, wv.z, wv.w};
        #pragma unroll
        for (int j = 0; j < 4; ++j) {
          __bf16 h = (__bf16)xs[j];
          Ahi[r * LDB + c + j] = h;
          Alo[r * LDB + c + j] = (__bf16)(xs[j] - (float)h);
          __bf16 hw = (__bf16)ws4[j];
          Bh[r * LDB + c + j] = hw;
          Bl[r * LDB + c + j] = (__bf16)(ws4[j] - (float)hw);
        }
      }
    }
    __syncthreads();                                // bf16 tiles ready

    // ---- build WMMA fragments ----
    // A (16-bit 16x32): lanes 0-15 -> M=l16, K={0..7,16..23}; lanes 16-31 -> K={8..15,24..31}
    v16bf ah[2], al[2];
    #pragma unroll
    for (int mi = 0; mi < 2; ++mi) {
      const int m  = waveM + mi * 16 + l16;
      const int k0 = hsel * 8;
      ah[mi] = concat8(*(const v8bf*)&Ahi[m * LDB + k0],
                       *(const v8bf*)&Ahi[m * LDB + k0 + 16]);
      al[mi] = concat8(*(const v8bf*)&Alo[m * LDB + k0],
                       *(const v8bf*)&Alo[m * LDB + k0 + 16]);
    }
    // B (32x16): lanes 0-15 -> N=l16, K={0..7,16..23}; lanes 16-31 -> N=l16, K={8..15,24..31}
    v16bf bh[4], bl[4];
    #pragma unroll
    for (int ni = 0; ni < 4; ++ni) {
      const int n  = waveN + ni * 16 + l16;
      const int k0 = hsel * 8;
      bh[ni] = concat8(*(const v8bf*)&Bh[n * LDB + k0],
                       *(const v8bf*)&Bh[n * LDB + k0 + 16]);
      bl[ni] = concat8(*(const v8bf*)&Bl[n * LDB + k0],
                       *(const v8bf*)&Bl[n * LDB + k0 + 16]);
    }

    // ---- 24 WMMAs: hi*hi + hi*lo + lo*hi (lo*lo negligible) ----
    #pragma unroll
    for (int mi = 0; mi < 2; ++mi) {
      #pragma unroll
      for (int ni = 0; ni < 4; ++ni) {
        v8f c = acc[mi * 4 + ni];
        c = wmma_bf16(ah[mi], bh[ni], c);
        c = wmma_bf16(ah[mi], bl[ni], c);
        c = wmma_bf16(al[mi], bh[ni], c);
        acc[mi * 4 + ni] = c;
      }
    }
    // no barrier needed here: next iteration's first barrier orders reuse
  }

  // ---- epilogue: C layout VGPR r -> M = r (+8 for lanes 16-31), N = l16 ----
  #pragma unroll
  for (int mi = 0; mi < 2; ++mi) {
    #pragma unroll
    for (int ni = 0; ni < 4; ++ni) {
      const int mrow = mBase + waveM + mi * 16 + hsel * 8;
      const int ncol = nBase + waveN + ni * 16 + l16;
      const v8f c = acc[mi * 4 + ni];
      #pragma unroll
      for (int r = 0; r < 8; ++r)
        Cp[(size_t)(mrow + r) * NST + ncol] = c[r];
    }
  }
}

// ---------------------------------------------------------------------------
// Kernel 2: inv_norm[m] = 1 / (||k[m,:]||_2 + eps), one wave32 per row.
// ---------------------------------------------------------------------------
__global__ __launch_bounds__(256)
void norm_kernel(const float* __restrict__ kbuf, float* __restrict__ invn) {
  const int row  = (blockIdx.x * blockDim.x + threadIdx.x) >> 5;
  const int lane = threadIdx.x & 31;
  const float4* r = (const float4*)(kbuf + (size_t)row * NST);
  float s = 0.0f;
  #pragma unroll
  for (int j = 0; j < 8; ++j) {
    const float4 v = r[lane + j * 32];
    s += v.x * v.x + v.y * v.y + v.z * v.z + v.w * v.w;
  }
  #pragma unroll
  for (int off = 16; off > 0; off >>= 1) s += __shfl_xor(s, off, 32);
  if (lane == 0) invn[row] = 1.0f / (sqrtf(s) + EPS_);
}

// ---------------------------------------------------------------------------
// Kernel 3: sequential scan over T, diagonal in (b,n). One thread per state
// element. Parallelism is capped at B*N = 16384 chains (~1 wave32/SIMD), so
// latency is hidden with a 16-deep rotating software pipeline (~48 loads in
// flight per wave, LOADcnt max is 63). q is staged in d_out and overwritten
// in place (read 16 steps ahead of the write).
// ---------------------------------------------------------------------------
__device__ __forceinline__ float fast_tanh(float x) {
#if __has_builtin(__builtin_amdgcn_tanhf)
  return __builtin_amdgcn_tanhf(x);
#elif __has_builtin(__builtin_amdgcn_tanh_f32)
  return __builtin_amdgcn_tanh_f32(x);
#else
  float r;
  asm("v_tanh_f32 %0, %1\n\ts_nop 1" : "=v"(r) : "v"(x));
  return r;
#endif
}

__device__ __forceinline__ float fast_sigmoid(float x) {
  const float e = __builtin_amdgcn_exp2f(-x * 1.4426950408889634f);
  return __builtin_amdgcn_rcpf(1.0f + e);
}

constexpr int SD = 16;                               // pipeline depth

__global__ __launch_bounds__(64)
void scan_kernel(const float* __restrict__ kbuf,
                 const float* __restrict__ vbuf,
                 const float* __restrict__ invn,
                 float* __restrict__ qout,           // q in, out written in place
                 float* __restrict__ sfinal) {
  const int b = blockIdx.x >> 4;                           // 0..15
  const int n = ((blockIdx.x & 15) << 6) + threadIdx.x;    // 0..1023
  const size_t colBase = (size_t)b * NST + n;
  const size_t stride  = (size_t)B_ * NST;                 // per-timestep slab

  float kk[SD], vv[SD], qq[SD], iv[SD];
  #pragma unroll
  for (int j = 0; j < SD; ++j) {                     // prologue: t = 0..SD-1
    const size_t idx = colBase + (size_t)j * stride;
    kk[j] = kbuf[idx];
    vv[j] = vbuf[idx];
    qq[j] = qout[idx];
    iv[j] = invn[j * B_ + b];
  }

  float S = 0.0f;
  for (int tb = 0; tb < T_ / SD; ++tb) {
    #pragma unroll
    for (int j = 0; j < SD; ++j) {
      const int t = tb * SD + j;
      const size_t cidx = colBase + (size_t)t * stride;
      const float kn = kk[j] * iv[j];
      const float Sn = fast_tanh(S * (1.0f - kn * kn) + vv[j] * kn);
      const float Sq = Sn * qq[j];
      qout[cidx] = Sq * Sq * fast_sigmoid(Sq);       // Sq * silu(Sq)
      S = Sn;
      if (t + SD < T_) {                             // refill slot j (uniform)
        const size_t nidx = cidx + (size_t)SD * stride;
        kk[j] = kbuf[nidx];
        vv[j] = vbuf[nidx];
        qq[j] = qout[nidx];
        iv[j] = invn[(t + SD) * B_ + b];
      }
    }
  }
  sfinal[colBase] = S;                               // second tuple output
}

// ---------------------------------------------------------------------------
// Launch: inputs in setup_inputs() order: x, W_k, W_v, W_q (all fp32).
// d_out = [output (T,B,N) | S_final (B,N)] fp32.
// d_ws  = [k (T*B*N) | v (T*B*N) | inv_norm (T*B)] fp32  (~268.5 MB needed).
// ---------------------------------------------------------------------------
extern "C" void kernel_launch(void* const* d_in, const int* in_sizes, int n_in,
                              void* d_out, int out_size, void* d_ws, size_t ws_size,
                              hipStream_t stream) {
  const float* x  = (const float*)d_in[0];
  const float* Wk = (const float*)d_in[1];
  const float* Wv = (const float*)d_in[2];
  const float* Wq = (const float*)d_in[3];

  float* out  = (float*)d_out;
  float* kbuf = (float*)d_ws;
  float* vbuf = kbuf + (size_t)M_TOTAL * NST;
  float* invn = vbuf + (size_t)M_TOTAL * NST;
  float* qbuf = out;                                 // stage q inside d_out
  float* sfin = out + (size_t)M_TOTAL * NST;         // S_final tail

  // dynamic LDS: 2 double-buffered fp32 staging tiles + 4 bf16 tiles
  const size_t ldsBytes = (size_t)(2 * FBUF + 2 * FBUF) * sizeof(float)
                        + (size_t)(4 * BM * LDB) * sizeof(__bf16);   // 114688 B

  dim3 gGemm(M_TOTAL / BM, 24);                      // 256 x 24 blocks
  gemm3_kernel<<<gGemm, dim3(256), ldsBytes, stream>>>(x, Wk, Wv, Wq,
                                                       kbuf, vbuf, qbuf);

  norm_kernel<<<dim3(M_TOTAL / 8), dim3(256), 0, stream>>>(kbuf, invn);

  scan_kernel<<<dim3(B_ * (NST / 64)), dim3(64), 0, stream>>>(kbuf, vbuf, invn,
                                                              qbuf, sfin);
}
